// BaichuanAttention_77446850281512
// MI455X (gfx1250) — compile-verified, matrix operand order checked by reading, not run on hardware
//
#include <hip/hip_runtime.h>
#include <math.h>

// ---------------------------------------------------------------------------
// Baichuan-13B prefill attention block for MI455X (gfx1250, wave32, WMMA).
//   K0: one-time f32 -> bf16 weight/activation conversion (makes w_pack ~L2
//       resident at 157 MB vs 192 MB L2, and GEMM main loops pure-bf16)
//   K1: QKV GEMM, async double-buffered LDS staging + wmma bf16 f32-accum
//   K2: ALiBi causal flash attention, async K/V^T tile streaming + wmma
//   K3: O-proj GEMM, same structure, f32 output
// ---------------------------------------------------------------------------

typedef __attribute__((ext_vector_type(8)))  __bf16 v8bf;
typedef __attribute__((ext_vector_type(16))) __bf16 v16bf;
typedef __attribute__((ext_vector_type(8)))  float  v8f;

#define HIDN  5120
#define NQKV  15360
#define SEQ   2048
#define NH    40
#define HD    128
#define NB    2

__device__ __forceinline__ v8f wmma_bf16(v16bf a, v16bf b, v8f c) {
  return __builtin_amdgcn_wmma_f32_16x16x32_bf16(false, a, false, b, (short)0, c,
                                                 false, false);
}

// A/B fragment gather matching the 16-bit WMMA register layout:
// lane half lh=0 holds K = [0..7] and [16..23]; lh=1 holds [8..15] and [24..31].
__device__ __forceinline__ v16bf ldsfrag(const __bf16* row, int lh) {
  v8bf lo = *(const v8bf*)(row + lh * 8);
  v8bf hi = *(const v8bf*)(row + lh * 8 + 16);
  return __builtin_shufflevector(lo, hi, 0, 1, 2, 3, 4, 5, 6, 7,
                                         8, 9, 10, 11, 12, 13, 14, 15);
}

__device__ __forceinline__ v8f zero8() {
  v8f z;
#pragma unroll
  for (int i = 0; i < 8; ++i) z[i] = 0.0f;
  return z;
}

// gfx1250 async copy: 16B per lane, global -> LDS, tracked by ASYNCcnt.
// LDS aperture address is addr[31:0], so a truncated __shared__ pointer is the
// DS byte offset expected in the VDST operand.
__device__ __forceinline__ void async_copy16(unsigned lds_off, const void* gsrc) {
  asm volatile("global_load_async_to_lds_b128 %0, %1, off"
               :
               : "v"(lds_off), "v"(gsrc)
               : "memory");
}
__device__ __forceinline__ void wait_asynccnt0() {
  asm volatile("s_wait_asynccnt 0" ::: "memory");
}

// ---------------------------------------------------------------------------
// Kernel 0: elementwise f32 -> bf16 (8 elements per thread).
// ---------------------------------------------------------------------------
__global__ __launch_bounds__(256) void cvt_bf16_kernel(
    const float* __restrict__ src, __bf16* __restrict__ dst, long n) {
  const long i = ((long)blockIdx.x * 256 + threadIdx.x) * 8;
  if (i + 8 > n) return;
  float4 a = *(const float4*)(src + i);
  float4 b = *(const float4*)(src + i + 4);
  v8bf p;
  p[0] = (__bf16)a.x; p[1] = (__bf16)a.y; p[2] = (__bf16)a.z; p[3] = (__bf16)a.w;
  p[4] = (__bf16)b.x; p[5] = (__bf16)b.y; p[6] = (__bf16)b.z; p[7] = (__bf16)b.w;
  *(v8bf*)(dst + i) = p;
}

// ---------------------------------------------------------------------------
// Kernel 1: QKV projection. C[t,n] = sum_k X[t,k] * W[n,k]  (TN, bf16 x bf16).
// 128x128 tile, K-step 32, async double-buffered LDS, 8 waves (2Mx4N).
// Epilogue scatters into Q [B,H,S,D], K [B,H,S,D], V^T [B,H,D,S] (bf16).
// ---------------------------------------------------------------------------
__global__ __launch_bounds__(256) void qkv_gemm_kernel(
    const __bf16* __restrict__ A, const __bf16* __restrict__ B,
    __bf16* __restrict__ qws, __bf16* __restrict__ kws,
    __bf16* __restrict__ vTws) {
  __shared__ __bf16 sA[2][128][40];  // stride 40 keeps 16B alignment (80B rows)
  __shared__ __bf16 sB[2][128][40];

  const int tid = threadIdx.x;
  const int lane = tid & 31, wave = tid >> 5;
  const int ln = lane & 15, lh = lane >> 4;
  const int m0 = blockIdx.y * 128, n0 = blockIdx.x * 128;
  const int wm = (wave & 1) * 64, wn = (wave >> 1) * 32;

  const unsigned sAoff = (unsigned)(size_t)&sA[0][0][0];
  const unsigned sBoff = (unsigned)(size_t)&sB[0][0][0];
  const unsigned BUFB = 128u * 40u * 2u;  // 10240 B per buffer

  // staging chunks: tile = 128 rows x 64B; 512 x 16B chunks; 2 per lane/operand
  int rowj[2], colj[2];
#pragma unroll
  for (int j = 0; j < 2; ++j) {
    const int g = wave * 64 + j * 32 + lane;
    rowj[j] = g >> 2;
    colj[j] = g & 3;
  }

  auto issue = [&](int buf, int k0) {
#pragma unroll
    for (int j = 0; j < 2; ++j) {
      const unsigned l = (unsigned)(rowj[j] * 80 + colj[j] * 16 + buf * (int)BUFB);
      async_copy16(sAoff + l, A + (size_t)(m0 + rowj[j]) * HIDN + k0 + colj[j] * 8);
      async_copy16(sBoff + l, B + (size_t)(n0 + rowj[j]) * HIDN + k0 + colj[j] * 8);
    }
  };

  v8f acc[4][2];
#pragma unroll
  for (int i = 0; i < 4; ++i)
#pragma unroll
    for (int j = 0; j < 2; ++j) acc[i][j] = zero8();

  const int NK = HIDN / 32;
  issue(0, 0);
  for (int ks = 0; ks < NK; ++ks) {
    const int cur = ks & 1;
    wait_asynccnt0();    // our async stores into buf[cur] are in LDS
    __syncthreads();     // everyone's stores done; buf[1-cur] reads done too
    if (ks + 1 < NK) issue(1 - cur, (ks + 1) * 32);
    if (ks + 2 < NK) {   // warm L2 two tiles ahead (global_prefetch_b8)
      __builtin_prefetch(A + (size_t)(m0 + rowj[0]) * HIDN + (ks + 2) * 32);
      __builtin_prefetch(B + (size_t)(n0 + rowj[0]) * HIDN + (ks + 2) * 32);
    }
    v16bf bf0 = ldsfrag(&sB[cur][wn + ln][0], lh);
    v16bf bf1 = ldsfrag(&sB[cur][wn + 16 + ln][0], lh);
#pragma unroll
    for (int mt = 0; mt < 4; ++mt) {
      v16bf af = ldsfrag(&sA[cur][wm + mt * 16 + ln][0], lh);
      acc[mt][0] = wmma_bf16(af, bf0, acc[mt][0]);
      acc[mt][1] = wmma_bf16(af, bf1, acc[mt][1]);
    }
  }

  // Epilogue: C lane layout is rows m = lh*8 + i (VGPR i), col n = ln.
#pragma unroll
  for (int mt = 0; mt < 4; ++mt) {
#pragma unroll
    for (int nt = 0; nt < 2; ++nt) {
#pragma unroll
      for (int i = 0; i < 8; ++i) {
        const int m = m0 + wm + mt * 16 + lh * 8 + i;
        const int n = n0 + wn + nt * 16 + ln;
        const __bf16 v = (__bf16)acc[mt][nt][i];
        const int b = m >> 11, s = m & 2047;  // SEQ = 2048
        const int d = n & 127;
        if (n < HIDN) {
          const int h = n >> 7;
          qws[(((size_t)(b * NH + h)) * SEQ + s) * HD + d] = v;
        } else if (n < 2 * HIDN) {
          const int h = (n - HIDN) >> 7;
          kws[(((size_t)(b * NH + h)) * SEQ + s) * HD + d] = v;
        } else {
          const int h = (n - 2 * HIDN) >> 7;
          vTws[(((size_t)(b * NH + h)) * HD + d) * SEQ + s] = v;
        }
      }
    }
  }
}

// ---------------------------------------------------------------------------
// Kernel 2: ALiBi causal flash attention.
// Grid: (S/128, B*H). 8 waves each own a 16-query tile; 64-key K / V^T tiles
// are async-streamed through double-buffered LDS; online softmax per row.
// ---------------------------------------------------------------------------
__global__ __launch_bounds__(256) void attn_kernel(
    const __bf16* __restrict__ qws, const __bf16* __restrict__ kws,
    const __bf16* __restrict__ vTws, __bf16* __restrict__ attnws) {
  __shared__ __bf16 sK[2][64][136];   // keys x head_dim (pad 8) : 2x17408 B
  __shared__ __bf16 sV[2][128][72];   // head_dim x keys (pad 8) : 2x18432 B
  __shared__ __bf16 sP[8][16][72];    // per-wave P tile (q x keys)

  const int tid = threadIdx.x;
  const int lane = tid & 31, wave = tid >> 5;
  const int ln = lane & 15, lh = lane >> 4;
  const int bh = blockIdx.y;
  const int b = bh / NH, h = bh % NH;
  const int q0blk = blockIdx.x * 128;
  const int qRow = q0blk + wave * 16;

  const float scale = 0.08838834764831845f;  // 1/sqrt(128)
  const float slope = (h < 32) ? exp2f(-0.25f * (float)(h + 1))
                               : exp2f(-0.125f * (float)(2 * (h - 32) + 1));

  const __bf16* Qbase = qws + ((size_t)(b * NH + h)) * SEQ * HD;
  const __bf16* Kbase = kws + ((size_t)(b * NH + h)) * SEQ * HD;
  const __bf16* Vbase = vTws + ((size_t)(b * NH + h)) * HD * SEQ;

  const unsigned sK0 = (unsigned)(size_t)&sK[0][0][0];
  const unsigned sV0 = (unsigned)(size_t)&sV[0][0][0];
  const unsigned KBUF = 64u * 136u * 2u;   // 17408
  const unsigned VBUF = 128u * 72u * 2u;   // 18432

  auto issue = [&](int buf, int kbase) {
#pragma unroll
    for (int j = 0; j < 4; ++j) {
      const int g = wave * 128 + j * 32 + lane;
      const int kr = g >> 4, kc = g & 15;      // K tile: 64 rows x 256B
      async_copy16(sK0 + buf * KBUF + kr * 272 + kc * 16,
                   Kbase + (size_t)(kbase + kr) * HD + kc * 8);
      const int vr = g >> 3, vc = g & 7;       // V^T tile: 128 rows x 128B
      async_copy16(sV0 + buf * VBUF + vr * 144 + vc * 16,
                   Vbase + (size_t)vr * SEQ + kbase + vc * 8);
    }
  };

  // Q fragments (A operand): lane row = ln, 4 K-chunks of 32 over D=128.
  v16bf qf[4];
  {
    const __bf16* qrow = Qbase + (size_t)(qRow + ln) * HD;
#pragma unroll
    for (int kc = 0; kc < 4; ++kc) qf[kc] = ldsfrag(qrow + kc * 32, lh);
  }

  v8f o[4];
#pragma unroll
  for (int dt = 0; dt < 4; ++dt) o[dt] = zero8();
  float mrow[8], lrow[8];
#pragma unroll
  for (int i = 0; i < 8; ++i) { mrow[i] = -INFINITY; lrow[i] = 0.0f; }

  const int nkb = (q0blk + 128) >> 6;  // causal key-block count for this WG
  issue(0, 0);
  for (int kb = 0; kb < nkb; ++kb) {
    const int cur = kb & 1;
    const int kbase = kb * 64;
    wait_asynccnt0();
    __syncthreads();
    if (kb + 1 < nkb) issue(1 - cur, (kb + 1) * 64);

    if (kbase <= qRow + 15) {  // block touches this wave's causal window
      // S = Q K^T : 4 key tiles x 4 D-chunks
      v8f sc[4];
#pragma unroll
      for (int ct = 0; ct < 4; ++ct) {
        sc[ct] = zero8();
#pragma unroll
        for (int kc = 0; kc < 4; ++kc) {
          v16bf kf = ldsfrag(&sK[cur][ct * 16 + ln][kc * 32], lh);
          sc[ct] = wmma_bf16(qf[kc], kf, sc[ct]);
        }
      }
      // scale + ALiBi + causal mask; per-row block max
      float bmax[8];
#pragma unroll
      for (int i = 0; i < 8; ++i) bmax[i] = -INFINITY;
#pragma unroll
      for (int ct = 0; ct < 4; ++ct) {
        const int kpos = kbase + ct * 16 + ln;
        const float bias = slope * (float)kpos;
#pragma unroll
        for (int i = 0; i < 8; ++i) {
          const int qpos = qRow + lh * 8 + i;
          float s = sc[ct][i] * scale + bias;
          s = (kpos <= qpos) ? s : -INFINITY;
          sc[ct][i] = s;
          bmax[i] = fmaxf(bmax[i], s);
        }
      }
#pragma unroll
      for (int i = 0; i < 8; ++i)
#pragma unroll
        for (int off = 1; off < 16; off <<= 1)
          bmax[i] = fmaxf(bmax[i], __shfl_xor(bmax[i], off, 32));

      // online softmax update
      float psum[8];
#pragma unroll
      for (int i = 0; i < 8; ++i) {
        const float mnew = fmaxf(mrow[i], bmax[i]);
        const float rs = (mrow[i] == mnew) ? 1.0f : __expf(mrow[i] - mnew);
        mrow[i] = mnew;
        lrow[i] *= rs;
#pragma unroll
        for (int dt = 0; dt < 4; ++dt) o[dt][i] *= rs;
        psum[i] = 0.0f;
      }
#pragma unroll
      for (int ct = 0; ct < 4; ++ct) {
#pragma unroll
        for (int i = 0; i < 8; ++i) {
          const float p = __expf(sc[ct][i] - mrow[i]);
          psum[i] += p;
          sP[wave][lh * 8 + i][ct * 16 + ln] = (__bf16)p;  // C->A lane transpose
        }
      }
#pragma unroll
      for (int i = 0; i < 8; ++i) {
#pragma unroll
        for (int off = 1; off < 16; off <<= 1)
          psum[i] += __shfl_xor(psum[i], off, 32);
        lrow[i] += psum[i];
      }

      // O += P V : 2 K-chunks over 64 keys x 4 D tiles
#pragma unroll
      for (int kc = 0; kc < 2; ++kc) {
        v16bf pf = ldsfrag(&sP[wave][ln][kc * 32], lh);
#pragma unroll
        for (int dt = 0; dt < 4; ++dt) {
          v16bf vf = ldsfrag(&sV[cur][dt * 16 + ln][kc * 32], lh);
          o[dt] = wmma_bf16(pf, vf, o[dt]);
        }
      }
    }
  }

  // normalize and write [T, HID] bf16
#pragma unroll
  for (int dt = 0; dt < 4; ++dt) {
#pragma unroll
    for (int i = 0; i < 8; ++i) {
      const int q = qRow + lh * 8 + i;
      const float val = o[dt][i] / lrow[i];
      attnws[(size_t)(b * SEQ + q) * HIDN + h * HD + dt * 16 + ln] = (__bf16)val;
    }
  }
}

// ---------------------------------------------------------------------------
// Kernel 3: output projection. out[t,n] = sum_k attn[t,k] * Wo[n,k], f32 out.
// ---------------------------------------------------------------------------
__global__ __launch_bounds__(256) void oproj_gemm_kernel(
    const __bf16* __restrict__ A, const __bf16* __restrict__ B,
    float* __restrict__ out) {
  __shared__ __bf16 sA[2][128][40];
  __shared__ __bf16 sB[2][128][40];

  const int tid = threadIdx.x;
  const int lane = tid & 31, wave = tid >> 5;
  const int ln = lane & 15, lh = lane >> 4;
  const int m0 = blockIdx.y * 128, n0 = blockIdx.x * 128;
  const int wm = (wave & 1) * 64, wn = (wave >> 1) * 32;

  const unsigned sAoff = (unsigned)(size_t)&sA[0][0][0];
  const unsigned sBoff = (unsigned)(size_t)&sB[0][0][0];
  const unsigned BUFB = 128u * 40u * 2u;

  int rowj[2], colj[2];
#pragma unroll
  for (int j = 0; j < 2; ++j) {
    const int g = wave * 64 + j * 32 + lane;
    rowj[j] = g >> 2;
    colj[j] = g & 3;
  }

  auto issue = [&](int buf, int k0) {
#pragma unroll
    for (int j = 0; j < 2; ++j) {
      const unsigned l = (unsigned)(rowj[j] * 80 + colj[j] * 16 + buf * (int)BUFB);
      async_copy16(sAoff + l, A + (size_t)(m0 + rowj[j]) * HIDN + k0 + colj[j] * 8);
      async_copy16(sBoff + l, B + (size_t)(n0 + rowj[j]) * HIDN + k0 + colj[j] * 8);
    }
  };

  v8f acc[4][2];
#pragma unroll
  for (int i = 0; i < 4; ++i)
#pragma unroll
    for (int j = 0; j < 2; ++j) acc[i][j] = zero8();

  const int NK = HIDN / 32;
  issue(0, 0);
  for (int ks = 0; ks < NK; ++ks) {
    const int cur = ks & 1;
    wait_asynccnt0();
    __syncthreads();
    if (ks + 1 < NK) issue(1 - cur, (ks + 1) * 32);
    if (ks + 2 < NK) {
      __builtin_prefetch(A + (size_t)(m0 + rowj[0]) * HIDN + (ks + 2) * 32);
      __builtin_prefetch(B + (size_t)(n0 + rowj[0]) * HIDN + (ks + 2) * 32);
    }
    v16bf bf0 = ldsfrag(&sB[cur][wn + ln][0], lh);
    v16bf bf1 = ldsfrag(&sB[cur][wn + 16 + ln][0], lh);
#pragma unroll
    for (int mt = 0; mt < 4; ++mt) {
      v16bf af = ldsfrag(&sA[cur][wm + mt * 16 + ln][0], lh);
      acc[mt][0] = wmma_bf16(af, bf0, acc[mt][0]);
      acc[mt][1] = wmma_bf16(af, bf1, acc[mt][1]);
    }
  }

#pragma unroll
  for (int mt = 0; mt < 4; ++mt) {
#pragma unroll
    for (int nt = 0; nt < 2; ++nt) {
#pragma unroll
      for (int i = 0; i < 8; ++i) {
        const int m = m0 + wm + mt * 16 + lh * 8 + i;
        const int n = n0 + wn + nt * 16 + ln;
        out[(size_t)m * HIDN + n] = acc[mt][nt][i];
      }
    }
  }
}

// ---------------------------------------------------------------------------
extern "C" void kernel_launch(void* const* d_in, const int* in_sizes, int n_in,
                              void* d_out, int out_size, void* d_ws,
                              size_t ws_size, hipStream_t stream) {
  (void)in_sizes; (void)n_in; (void)out_size; (void)ws_size;
  const float* hidden = (const float*)d_in[0];
  const float* w_pack = (const float*)d_in[1];
  const float* o_proj = (const float*)d_in[2];
  // d_in[3..5] (k_cache, v_cache, block_offsets): cache fill + gather through
  // the block table is a permutation covering exactly the fresh K/V, so the
  // net dataflow is identity; we keep K / V^T in workspace (inputs immutable).
  float* out = (float*)d_out;

  const size_t nX = (size_t)NB * SEQ * HIDN;   // 20,971,520
  const size_t nW = (size_t)NQKV * HIDN;       // 78,643,200
  const size_t nO = (size_t)HIDN * HIDN;       // 26,214,400
  const size_t per = (size_t)NB * NH * SEQ * HD;

  __bf16* xbf    = (__bf16*)d_ws;
  __bf16* wqkv   = xbf + nX;
  __bf16* wo     = wqkv + nW;
  __bf16* qws    = wo + nO;
  __bf16* kws    = qws + per;
  __bf16* vTws   = kws + per;
  __bf16* attnws = vTws + per;  // [T, HID] bf16

  cvt_bf16_kernel<<<(unsigned)(nX / 8 / 256), 256, 0, stream>>>(hidden, xbf, (long)nX);
  cvt_bf16_kernel<<<(unsigned)(nW / 8 / 256), 256, 0, stream>>>(w_pack, wqkv, (long)nW);
  cvt_bf16_kernel<<<(unsigned)(nO / 8 / 256), 256, 0, stream>>>(o_proj, wo, (long)nO);

  qkv_gemm_kernel<<<dim3(NQKV / 128, (NB * SEQ) / 128), 256, 0, stream>>>(
      xbf, wqkv, qws, kws, vTws);
  attn_kernel<<<dim3(SEQ / 128, NB * NH), 256, 0, stream>>>(
      qws, kws, vTws, attnws);
  oproj_gemm_kernel<<<dim3(HIDN / 128, (NB * SEQ) / 128), 256, 0, stream>>>(
      attnws, wo, out);
}
